// Generator_3470333575609
// MI455X (gfx1250) — matrix operand order repaired and checked by reading, not compiled, not run on hardware
//
#include <hip/hip_runtime.h>

// Problem constants (match reference)
#define B_ 8192
#define E_ 128
#define H_ 512
#define V_ 128
#define G_ 2048   // 4*H

typedef _Float16 half_t;
typedef _Float16 v16h __attribute__((ext_vector_type(16)));
typedef _Float16 v8h  __attribute__((ext_vector_type(8)));
typedef float    v8f  __attribute__((ext_vector_type(8)));
typedef int      v4i  __attribute__((ext_vector_type(4)));

// ---- CDNA5 async global->LDS path (probe-verified on this toolchain) ----
#if defined(__HIP_DEVICE_COMPILE__) && \
    __has_builtin(__builtin_amdgcn_global_load_async_to_lds_b128) && \
    __has_builtin(__builtin_amdgcn_s_wait_asynccnt)
#define USE_ASYNC_LDS 1
typedef __attribute__((address_space(1))) v4i gv4i_t;   // global int4
typedef __attribute__((address_space(3))) v4i lv4i_t;   // LDS int4
#endif

// Scheduler fence: force "all fragment loads, then all WMMAs" ordering.
#if defined(__HIP_DEVICE_COMPILE__) && __has_builtin(__builtin_amdgcn_sched_barrier)
#define SCHED_FENCE() __builtin_amdgcn_sched_barrier(0)
#else
#define SCHED_FENCE()
#endif

__device__ __forceinline__ void copy16(const half_t* g, half_t* l) {
#ifdef USE_ASYNC_LDS
    __builtin_amdgcn_global_load_async_to_lds_b128((gv4i_t*)g, (lv4i_t*)l, 0, 0);
#else
    *(v8h*)l = *(const v8h*)g;
#endif
}

__device__ __forceinline__ void wait_stage() {
#ifdef USE_ASYNC_LDS
    __builtin_amdgcn_s_wait_asynccnt(0);
#endif
}

// ---------------- prep kernels ----------------

__global__ __launch_bounds__(256)
void convert_f32_to_f16(const float* __restrict__ in, half_t* __restrict__ out, int n) {
    int i = blockIdx.x * blockDim.x + threadIdx.x;
    if (i < n) out[i] = (half_t)in[i];
}

// in: [N][K] row-major (fp32)  ->  out: [K][N] (f16)
__global__ __launch_bounds__(256)
void convert_transpose(const float* __restrict__ in, half_t* __restrict__ out, int N, int K) {
    int i = blockIdx.x * blockDim.x + threadIdx.x;
    if (i < N * K) {
        int n = i / K, k = i - n * K;
        out[(size_t)k * N + n] = (half_t)in[i];
    }
}

// X0[b][e] = (f16) embed_W[x[b]][e]
__global__ __launch_bounds__(256)
void embed_kernel(const int* __restrict__ x, const float* __restrict__ embed_W,
                  half_t* __restrict__ X0) {
    int i = blockIdx.x * blockDim.x + threadIdx.x;   // over B*E
    int b = i >> 7, e = i & (E_ - 1);
    X0[i] = (half_t)embed_W[(size_t)x[b] * E_ + e];
}

// ---------------- tiled WMMA GEMM phase ----------------
// Block tile: 128 rows (batch) x 32 cols, replicated over NG "gates"
// (gate g columns live at  n0 + g*gateStride  in WT's N dimension).
// 8 waves; wave wm owns rows wm*16..wm*16+15, all 32 cols.
// BK=64 staging (two 32-K sub-steps per buffer) -> NG*4 WMMAs per wave per
// barrier. Double-buffered LDS: sA[2][128][72] f16, sB[2][NG][64][32] f16.

#define SA_BUF (128 * 72)

template <int NG>
__device__ __forceinline__ void stage_tiles(
    const half_t* __restrict__ A, int ldA, int k0, int blockRow,
    const half_t* __restrict__ WT, int ldW, int n0, int gateStride,
    half_t* __restrict__ sA, half_t* __restrict__ sB)
{
    const int tid = threadIdx.x;
    // A tile: 128 rows x 64 halves = 1024 b128 tasks (4 per thread)
#pragma unroll
    for (int t = 0; t < 1024; t += 256) {
        int u = t + tid;
        int row = u >> 3, ch = u & 7;
        copy16(A + (size_t)(blockRow + row) * ldA + k0 + ch * 8, sA + row * 72 + ch * 8);
    }
    // B tiles: NG x 64(k) x 32(n) halves = NG*256 b128 tasks
#pragma unroll
    for (int t = 0; t < NG * 256; t += 256) {
        int u = t + tid;
        int g = u >> 8, rem = u & 255;
        int kk = rem >> 2, ch = rem & 3;
        copy16(WT + (size_t)(k0 + kk) * ldW + n0 + g * gateStride + ch * 8,
               sB + g * 2048 + kk * 32 + ch * 8);
    }
}

template <int NG>
__device__ __forceinline__ void gemm_phase(
    const half_t* __restrict__ A, int ldA, int K,
    const half_t* __restrict__ WT, int ldW,
    int blockRow, int n0, int gateStride,
    half_t* __restrict__ sA, half_t* __restrict__ sB, v8f (*acc)[2])
{
    const int tid  = threadIdx.x;
    const int lane = tid & 31;
    const int wm   = tid >> 5;         // 0..7 : wave's 16-row band
    const int r    = lane & 15;
    const int klo  = (lane >> 4) ? 8 : 0;
    const int S    = K / 64;           // K in {128,512} -> S in {2,8}

    stage_tiles<NG>(A, ldA, 0, blockRow, WT, ldW, n0, gateStride, sA, sB);

    for (int s = 0; s < S; ++s) {
        wait_stage();                  // this wave's async stores for buf s&1 done
        __syncthreads();               // tiles visible; prior reads of buf (s+1)&1 done
        if (s + 1 < S)
            stage_tiles<NG>(A, ldA, (s + 1) * 64, blockRow, WT, ldW, n0, gateStride,
                            sA + ((s + 1) & 1) * SA_BUF,
                            sB + ((s + 1) & 1) * (NG * 2048));
        const half_t* cA = sA + (s & 1) * SA_BUF;
        const half_t* cB = sB + (s & 1) * (NG * 2048);

#pragma unroll
        for (int ks = 0; ks < 64; ks += 32) {
            // A fragment (16x32 f16): lane r, K chunks {klo..+7, 16+klo..+7}
            v16h afv;
            {
                union { v16h v; v8h h[2]; } t;
                const half_t* ap = cA + (wm * 16 + r) * 72 + ks + klo;
                t.h[0] = *(const v8h*)(ap);
                t.h[1] = *(const v8h*)(ap + 16);
                afv = t.v;
            }
            // preload ALL B fragments ...
            v16h bfv[NG][2];
#pragma unroll
            for (int g = 0; g < NG; ++g)
#pragma unroll
                for (int nt = 0; nt < 2; ++nt) {
                    const half_t* bp = cB + g * 2048 + (ks + klo) * 32 + nt * 16 + r;
#pragma unroll
                    for (int i = 0; i < 8; ++i) {
                        bfv[g][nt][i]     = bp[i * 32];
                        bfv[g][nt][8 + i] = bp[(16 + i) * 32];
                    }
                }
            SCHED_FENCE();   // ... then fire the WMMA chain back-to-back
#pragma unroll
            for (int g = 0; g < NG; ++g)
#pragma unroll
                for (int nt = 0; nt < 2; ++nt)
                    acc[g][nt] = __builtin_amdgcn_wmma_f32_16x16x32_f16(
                        false, afv, false, bfv[g][nt], (short)0, acc[g][nt], false, false);
            SCHED_FENCE();
        }
    }
    __syncthreads();                   // safe buffer reuse by the next phase
}

// ---------------- fused LSTM layer: gates GEMM + cell ----------------
__global__ __launch_bounds__(256)
void lstm_layer_kernel(const half_t* __restrict__ Aih, int Kih,
                       const half_t* __restrict__ WihT,   // [Kih][2048]
                       const half_t* __restrict__ Ahh,    // [B][512] f16 h state
                       const half_t* __restrict__ WhhT,   // [512][2048]
                       const float* __restrict__ bih, const float* __restrict__ bhh,
                       const float* __restrict__ c_in,    // [B][512]
                       float* __restrict__ hs_out, float* __restrict__ cs_out,
                       half_t* __restrict__ hf_out)
{
    __shared__ __attribute__((aligned(16))) half_t sA[2 * SA_BUF];
    __shared__ __attribute__((aligned(16))) half_t sB[2 * 4 * 2048];

    const int blockRow = blockIdx.x * 128;
    const int n0       = blockIdx.y * 32;       // column base within H

    v8f acc[4][2] = {};
    gemm_phase<4>(Aih, Kih, Kih, WihT, G_, blockRow, n0, H_, sA, sB, acc);
    gemm_phase<4>(Ahh, H_,  H_,  WhhT, G_, blockRow, n0, H_, sA, sB, acc);

    // epilogue: C/D layout -> (M = rr + 8*(lane>=16), N = lane&15)
    const int lane = threadIdx.x & 31;
    const int wm   = threadIdx.x >> 5;
    const int rowBase = blockRow + wm * 16 + ((lane >> 4) << 3);

#pragma unroll
    for (int nt = 0; nt < 2; ++nt) {
        const int col = n0 + nt * 16 + (lane & 15);
        const float bi = bih[col]        + bhh[col];
        const float bf = bih[col +  512] + bhh[col +  512];
        const float bg = bih[col + 1024] + bhh[col + 1024];
        const float bo = bih[col + 1536] + bhh[col + 1536];
#pragma unroll
        for (int rr = 0; rr < 8; ++rr) {
            const size_t idx = (size_t)(rowBase + rr) * H_ + col;
            float iv = 1.f / (1.f + __expf(-(acc[0][nt][rr] + bi)));
            float fv = 1.f / (1.f + __expf(-(acc[1][nt][rr] + bf)));
            float gv = tanhf(acc[2][nt][rr] + bg);
            float ov = 1.f / (1.f + __expf(-(acc[3][nt][rr] + bo)));
            float c2 = fv * c_in[idx] + iv * gv;
            float h2 = ov * tanhf(c2);
            cs_out[idx] = c2;
            hs_out[idx] = h2;
            hf_out[idx] = (half_t)h2;
        }
    }
}

// ---------------- output head: logits = h2 @ lin_W^T + b ----------------
__global__ __launch_bounds__(256)
void linear_kernel(const half_t* __restrict__ A,      // [B][512] f16
                   const half_t* __restrict__ WT,     // [512][128]
                   const float* __restrict__ bias,    // [128]
                   float* __restrict__ out)           // [B][128]
{
    __shared__ __attribute__((aligned(16))) half_t sA[2 * SA_BUF];
    __shared__ __attribute__((aligned(16))) half_t sB[2 * 2048];

    const int blockRow = blockIdx.x * 128;
    const int n0       = blockIdx.y * 32;

    v8f acc[1][2] = {};
    gemm_phase<1>(A, H_, H_, WT, V_, blockRow, n0, 0, sA, sB, acc);

    const int lane = threadIdx.x & 31;
    const int wm   = threadIdx.x >> 5;
    const int rowBase = blockRow + wm * 16 + ((lane >> 4) << 3);

#pragma unroll
    for (int nt = 0; nt < 2; ++nt) {
        const int col = n0 + nt * 16 + (lane & 15);
        const float bv = bias[col];
#pragma unroll
        for (int rr = 0; rr < 8; ++rr)
            out[(size_t)(rowBase + rr) * V_ + col] = acc[0][nt][rr] + bv;
    }
}

// ---------------- host side ----------------
extern "C" void kernel_launch(void* const* d_in, const int* in_sizes, int n_in,
                              void* d_out, int out_size, void* d_ws, size_t ws_size,
                              hipStream_t stream) {
    const int*   x       = (const int*)  d_in[0];
    const float* h_in    = (const float*)d_in[1];   // [3,B,H]
    const float* c_in    = (const float*)d_in[2];   // [3,B,H]
    const float* embed_W = (const float*)d_in[3];
    const float* Wih0 = (const float*)d_in[4];
    const float* Whh0 = (const float*)d_in[5];
    const float* bih0 = (const float*)d_in[6];
    const float* bhh0 = (const float*)d_in[7];
    const float* Wih1 = (const float*)d_in[8];
    const float* Whh1 = (const float*)d_in[9];
    const float* bih1 = (const float*)d_in[10];
    const float* bhh1 = (const float*)d_in[11];
    const float* Wih2 = (const float*)d_in[12];
    const float* Whh2 = (const float*)d_in[13];
    const float* bih2 = (const float*)d_in[14];
    const float* bhh2 = (const float*)d_in[15];
    const float* lin_W = (const float*)d_in[16];
    const float* lin_b = (const float*)d_in[17];

    float* logits = (float*)d_out;                       // [B,V]
    float* hs     = logits + (size_t)B_ * V_;            // [3,B,H]
    float* cs     = hs + 3ull * B_ * H_;                 // [3,B,H]

    // workspace carve (f16), 256B-aligned slices
    half_t* w = (half_t*)d_ws;
    size_t off = 0;
    auto alloc = [&](size_t n) { half_t* p = w + off; off += (n + 127) & ~(size_t)127; return p; };
    half_t* WihT0 = alloc((size_t)128 * G_);
    half_t* WihT1 = alloc((size_t)512 * G_);
    half_t* WihT2 = alloc((size_t)512 * G_);
    half_t* WhhT0 = alloc((size_t)512 * G_);
    half_t* WhhT1 = alloc((size_t)512 * G_);
    half_t* WhhT2 = alloc((size_t)512 * G_);
    half_t* linT  = alloc((size_t)512 * V_);
    half_t* X0    = alloc((size_t)B_ * E_);
    half_t* HfIn  = alloc(3ull * B_ * H_);               // f16 of input h, all layers
    half_t* Hf0   = alloc((size_t)B_ * H_);
    half_t* Hf1   = alloc((size_t)B_ * H_);
    half_t* Hf2   = alloc((size_t)B_ * H_);

    // --- prep: weight transpose+convert, state convert, embedding gather ---
    convert_transpose<<<(G_ * 128) / 256, 256, 0, stream>>>(Wih0, WihT0, G_, 128);
    convert_transpose<<<(G_ * 512) / 256, 256, 0, stream>>>(Whh0, WhhT0, G_, 512);
    convert_transpose<<<(G_ * 512) / 256, 256, 0, stream>>>(Wih1, WihT1, G_, 512);
    convert_transpose<<<(G_ * 512) / 256, 256, 0, stream>>>(Whh1, WhhT1, G_, 512);
    convert_transpose<<<(G_ * 512) / 256, 256, 0, stream>>>(Wih2, WihT2, G_, 512);
    convert_transpose<<<(G_ * 512) / 256, 256, 0, stream>>>(Whh2, WhhT2, G_, 512);
    convert_transpose<<<(V_ * 512) / 256, 256, 0, stream>>>(lin_W, linT, V_, 512);
    convert_f32_to_f16<<<(3 * B_ * H_) / 256, 256, 0, stream>>>(h_in, HfIn, 3 * B_ * H_);
    embed_kernel<<<(B_ * E_) / 256, 256, 0, stream>>>(x, embed_W, X0);

    // --- 3 fused LSTM layers (sequential dependency on stream) ---
    dim3 gridL(B_ / 128, H_ / 32);
    lstm_layer_kernel<<<gridL, 256, 0, stream>>>(
        X0, 128, WihT0, HfIn, WhhT0, bih0, bhh0,
        c_in, hs, cs, Hf0);
    lstm_layer_kernel<<<gridL, 256, 0, stream>>>(
        Hf0, 512, WihT1, HfIn + (size_t)B_ * H_, WhhT1, bih1, bhh1,
        c_in + (size_t)B_ * H_, hs + (size_t)B_ * H_, cs + (size_t)B_ * H_, Hf1);
    lstm_layer_kernel<<<gridL, 256, 0, stream>>>(
        Hf1, 512, WihT2, HfIn + 2ull * B_ * H_, WhhT2, bih2, bhh2,
        c_in + 2ull * B_ * H_, hs + 2ull * B_ * H_, cs + 2ull * B_ * H_, Hf2);

    // --- output head ---
    linear_kernel<<<dim3(B_ / 128, V_ / 32), 256, 0, stream>>>(Hf2, linT, lin_b, logits);
}
